// RelationalMemoryAdapter_8529805049879
// MI455X (gfx1250) — compile-verified, matrix operand order checked by reading, not run on hardware
//
#include <hip/hip_runtime.h>

// MI455X / gfx1250: wave32, WMMA 16x16x32 bf16 -> f32.
typedef __bf16  bf16x8  __attribute__((ext_vector_type(8)));
typedef __bf16  bf16x16 __attribute__((ext_vector_type(16)));
typedef float   v4f     __attribute__((ext_vector_type(4)));
typedef float   v8f     __attribute__((ext_vector_type(8)));

constexpr int B_ = 8, S_ = 2048, D_ = 2048, M_ = 128;
constexpr int WAVES = 4;                       // 4 wave32 per workgroup (one per SIMD32)
constexpr float SCALE = 0.022097086912079608f; // 1/sqrt(2048)

// Padded LDS row strides (bank-conflict avoidance, keep 16B alignment):
constexpr int KPAD = 40;   // 32 bf16 + 8 pad  (80 B rows)
constexpr int VPAD = 136;  // 128 bf16 + 8 pad (272 B rows)
constexpr int APAD = 132;  // 128 f32 + 4 pad  (528 B rows)

// sched_group_barrier masks
#define SGB_WMMA   0x008   // MFMA/WMMA class
#define SGB_DSREAD 0x100   // DS read

__global__ __launch_bounds__(WAVES * 32)
void relmem_attn_kernel(const float* __restrict__ hs,
                        const float* __restrict__ mt,
                        const unsigned char* __restrict__ msk,
                        float* __restrict__ out)
{
    // Double-buffered staging (one barrier per loop iteration, global prefetch
    // overlaps the WMMA stream of the previous chunk).
    __shared__ alignas(16) __bf16 Kbuf[2][M_][KPAD];     // K chunk  [m][k] bf16
    __shared__ alignas(16) __bf16 Vbuf[2][16][VPAD];     // V tile   [d][m] bf16 (transposed)
    __shared__ alignas(16) float  Attn[WAVES][16][APAD]; // attn     [wave][row][m]

    const int tid  = threadIdx.x;
    const int w    = tid >> 5;         // wave id 0..3
    const int lane = tid & 31;
    const int h    = lane >> 4;        // half-wave
    const int nn   = lane & 15;        // A: row index; B/C: column index

    const int b     = blockIdx.y;
    const int qbase = blockIdx.x * (16 * WAVES) + w * 16;

    const float* Qw = hs + ((size_t)b * S_ + qbase) * (size_t)D_;
    const float* Kb = mt + (size_t)b * M_ * (size_t)D_;
    const unsigned char* mb = msk + b * M_;

    // Per-lane mask bytes for m = 16t + nn, plus batch-valid flag.
    int mval[8]; int anyv = 0;
    #pragma unroll
    for (int t = 0; t < 8; ++t) { mval[t] = (int)mb[t * 16 + nn]; anyv |= mval[t]; }
    #pragma unroll
    for (int off = 1; off <= 8; off <<= 1) anyv |= __shfl_xor(anyv, off, 32);

    v8f acc[8];
    #pragma unroll
    for (int t = 0; t < 8; ++t) acc[t] = v8f{0.f,0.f,0.f,0.f,0.f,0.f,0.f,0.f};

    // ------------------- Pass A: scores = Q . K^T -------------------
    const int sub = tid & 7;     // k sub-group (4 floats)
    const int rb8 = tid >> 3;    // row base
    v4f kpre[8];
    #pragma unroll
    for (int r = 0; r < 8; ++r)
        kpre[r] = *(const v4f*)(Kb + (size_t)(r * 16 + rb8) * D_ + sub * 4);

    #pragma unroll 1
    for (int kc = 0; kc < D_ / 32; ++kc) {
        const int k0  = kc * 32;
        const int cur = kc & 1;
        // Commit prefetched K chunk -> bf16 LDS (shared by 4 waves)
        #pragma unroll
        for (int r = 0; r < 8; ++r) {
            const int m = r * 16 + rb8;
            __bf16* dst = &Kbuf[cur][m][sub * 4];
            dst[0] = (__bf16)kpre[r][0]; dst[1] = (__bf16)kpre[r][1];
            dst[2] = (__bf16)kpre[r][2]; dst[3] = (__bf16)kpre[r][3];
        }
        // Prefetch next K chunk from global (overlaps barrier + WMMAs)
        if (kc + 1 < D_ / 32) {
            #pragma unroll
            for (int r = 0; r < 8; ++r)
                kpre[r] = *(const v4f*)(Kb + (size_t)(r * 16 + rb8) * D_
                                        + k0 + 32 + sub * 4);
        }
        // Q fragment global loads (independent of LDS; issue before barrier)
        const float* qp = Qw + (size_t)nn * D_ + k0 + h * 8;
        v4f q0 = *(const v4f*)(qp);
        v4f q1 = *(const v4f*)(qp + 4);
        v4f q2 = *(const v4f*)(qp + 16);
        v4f q3 = *(const v4f*)(qp + 20);

        __syncthreads();

        // A fragment: 16-bit A layout (lane%16 = row; k runs 8h..8h+7, 16+8h..)
        bf16x16 afrag;
        #pragma unroll
        for (int i = 0; i < 4; ++i) {
            afrag[i]      = (__bf16)q0[i];
            afrag[4 + i]  = (__bf16)q1[i];
            afrag[8 + i]  = (__bf16)q2[i];
            afrag[12 + i] = (__bf16)q3[i];
        }
        // 8 memory-slot tiles: B[k][c] = K[16t+c][k0+k].
        #pragma unroll
        for (int t = 0; t < 8; ++t) {
            const int m = t * 16 + nn;
            bf16x8 lo = *(const bf16x8*)&Kbuf[cur][m][h * 8];
            bf16x8 hi = *(const bf16x8*)&Kbuf[cur][m][16 + h * 8];
            bf16x16 bfrag = __builtin_shufflevector(lo, hi,
                0,1,2,3,4,5,6,7,8,9,10,11,12,13,14,15);
            acc[t] = __builtin_amdgcn_wmma_f32_16x16x32_bf16(
                false, afrag, false, bfrag, (short)0, acc[t], false, false);
        }
        // Pin schedule: stay one B-fragment (2 ds_load_b128) ahead of each WMMA
        // so DS latency hides under the matrix pipe.
        __builtin_amdgcn_sched_group_barrier(SGB_DSREAD, 4, 0);
        __builtin_amdgcn_sched_group_barrier(SGB_WMMA,   1, 0);
        #pragma unroll
        for (int g = 0; g < 6; ++g) {
            __builtin_amdgcn_sched_group_barrier(SGB_DSREAD, 2, 0);
            __builtin_amdgcn_sched_group_barrier(SGB_WMMA,   1, 0);
        }
        __builtin_amdgcn_sched_group_barrier(SGB_WMMA, 1, 0);
    }

    // ------------------- Softmax over m (per row) -------------------
    // C layout: element j of acc[t] = score(row 8h+j, col 16t+nn).
    #pragma unroll
    for (int j = 0; j < 8; ++j) {
        float sv[8]; float mx = -3.0e38f;
        #pragma unroll
        for (int t = 0; t < 8; ++t) {
            float s = acc[t][j] * SCALE;
            s = mval[t] ? s : -1.0e30f;
            sv[t] = s; mx = fmaxf(mx, s);
        }
        #pragma unroll
        for (int off = 1; off <= 8; off <<= 1) mx = fmaxf(mx, __shfl_xor(mx, off, 32));
        float sum = 0.f;
        #pragma unroll
        for (int t = 0; t < 8; ++t) { sv[t] = __expf(sv[t] - mx); sum += sv[t]; }
        #pragma unroll
        for (int off = 1; off <= 8; off <<= 1) sum += __shfl_xor(sum, off, 32);
        const float inv = anyv ? (1.f / sum) : 0.f;  // zero all-masked batches
        #pragma unroll
        for (int t = 0; t < 8; ++t) Attn[w][8 * h + j][t * 16 + nn] = sv[t] * inv;
    }
    __syncthreads();

    // attn -> 4 bf16 A fragments (K dim = m, chunks of 32), kept in VGPRs.
    bf16x16 af[4];
    #pragma unroll
    for (int t2 = 0; t2 < 4; ++t2) {
        const int mb2 = 32 * t2 + 8 * h;
        const float* ap = &Attn[w][nn][0];
        v4f a0 = *(const v4f*)(ap + mb2);
        v4f a1 = *(const v4f*)(ap + mb2 + 4);
        v4f a2 = *(const v4f*)(ap + mb2 + 16);
        v4f a3 = *(const v4f*)(ap + mb2 + 20);
        #pragma unroll
        for (int i = 0; i < 4; ++i) {
            af[t2][i]      = (__bf16)a0[i];
            af[t2][4 + i]  = (__bf16)a1[i];
            af[t2][8 + i]  = (__bf16)a2[i];
            af[t2][12 + i] = (__bf16)a3[i];
        }
    }

    // ------------------- Pass B: out = attn . V -------------------
    const int c4  = tid & 3;
    const int rb4 = tid >> 2;
    float* outw = out + ((size_t)b * S_ + qbase) * (size_t)D_;

    v4f vpre[4];
    #pragma unroll
    for (int r = 0; r < 4; ++r)
        vpre[r] = *(const v4f*)(Kb + (size_t)(r * 32 + rb4) * D_ + c4 * 4);

    #pragma unroll 1
    for (int nt = 0; nt < D_ / 16; ++nt) {
        const int d0  = nt * 16;
        const int cur = nt & 1;
        // Commit prefetched V tile transposed -> Vbuf[cur][d][m] bf16
        #pragma unroll
        for (int r = 0; r < 4; ++r) {
            const int m = r * 32 + rb4;
            Vbuf[cur][c4 * 4 + 0][m] = (__bf16)vpre[r][0];
            Vbuf[cur][c4 * 4 + 1][m] = (__bf16)vpre[r][1];
            Vbuf[cur][c4 * 4 + 2][m] = (__bf16)vpre[r][2];
            Vbuf[cur][c4 * 4 + 3][m] = (__bf16)vpre[r][3];
        }
        // Prefetch next V tile (overlaps barrier + WMMAs + stores)
        if (nt + 1 < D_ / 16) {
            #pragma unroll
            for (int r = 0; r < 4; ++r)
                vpre[r] = *(const v4f*)(Kb + (size_t)(r * 32 + rb4) * D_
                                        + d0 + 16 + c4 * 4);
        }
        __syncthreads();

        v8f oacc = v8f{0.f,0.f,0.f,0.f,0.f,0.f,0.f,0.f};
        #pragma unroll
        for (int t2 = 0; t2 < 4; ++t2) {
            const int mb2 = 32 * t2 + 8 * h;
            bf16x8 lo = *(const bf16x8*)&Vbuf[cur][nn][mb2];
            bf16x8 hi = *(const bf16x8*)&Vbuf[cur][nn][mb2 + 16];
            bf16x16 bfrag = __builtin_shufflevector(lo, hi,
                0,1,2,3,4,5,6,7,8,9,10,11,12,13,14,15);
            oacc = __builtin_amdgcn_wmma_f32_16x16x32_bf16(
                false, af[t2], false, bfrag, (short)0, oacc, false, false);
        }
        // Pin schedule: one fragment ahead of each WMMA.
        __builtin_amdgcn_sched_group_barrier(SGB_DSREAD, 4, 0);
        __builtin_amdgcn_sched_group_barrier(SGB_WMMA,   1, 0);
        __builtin_amdgcn_sched_group_barrier(SGB_DSREAD, 2, 0);
        __builtin_amdgcn_sched_group_barrier(SGB_WMMA,   1, 0);
        __builtin_amdgcn_sched_group_barrier(SGB_DSREAD, 2, 0);
        __builtin_amdgcn_sched_group_barrier(SGB_WMMA,   2, 0);

        // C layout store: row 8h+j, column d0+nn (coalesced 64B across lanes 0..15)
        float* op = outw + d0 + nn;
        #pragma unroll
        for (int j = 0; j < 8; ++j) op[(size_t)(8 * h + j) * D_] = oacc[j];
    }
}

extern "C" void kernel_launch(void* const* d_in, const int* in_sizes, int n_in,
                              void* d_out, int out_size, void* d_ws, size_t ws_size,
                              hipStream_t stream) {
    const float*         hs = (const float*)d_in[0];          // [8,2048,2048]
    const float*         mt = (const float*)d_in[1];          // [8,128,2048]
    const unsigned char* mk = (const unsigned char*)d_in[2];  // [8,128] bool
    float* out = (float*)d_out;                               // [8,2048,2048]

    dim3 grid(S_ / (16 * WAVES), B_);   // (32 query tiles, 8 batches)
    dim3 block(WAVES * 32);             // 4 wave32
    hipLaunchKernelGGL(relmem_attn_kernel, grid, block, 0, stream, hs, mt, mk, out);
}
